// LSTM_38663295598629
// MI455X (gfx1250) — matrix operand order repaired and checked by reading, not compiled, not run on hardware
//
#include <hip/hip_runtime.h>
#include <hip/hip_bf16.h>
#include <math.h>

// Problem constants (from reference): B=64, T=512, I=1024, H=1024
#define BB 64
#define TT 512
#define II 1024
#define HH 1024
#define GG 4096   // 4*H packed gates: f, w, i, m

typedef __attribute__((ext_vector_type(16))) _Float16 v16h;
typedef __attribute__((ext_vector_type(8)))  _Float16 h8;
typedef __attribute__((ext_vector_type(4)))  _Float16 h4;
typedef __attribute__((ext_vector_type(8)))  float    v8f;
typedef __attribute__((ext_vector_type(4)))  float    f4;

// ---------------------------------------------------------------------------
// Fragment builders (layouts per CDNA5 ISA 7.12.2, wave32)
// ---------------------------------------------------------------------------
__device__ __forceinline__ v16h make_frag(h8 lo, h8 hi) {
  v16h r;
#pragma unroll
  for (int i = 0; i < 8; ++i) { r[i] = lo[i]; r[8 + i] = hi[i]; }
  return r;
}

// A-matrix 16x32 f16 tile at (m0,k0) from row-major fp16 matrix, leading dim lda.
// lanes 0-15: row=lane,  K chunks [k0+0..7] and [k0+16..23]
// lanes16-31: row=lane-16,K chunks [k0+8..15] and [k0+24..31]
__device__ __forceinline__ v16h load_a_f16(const _Float16* __restrict__ A, int lda,
                                           int m0, int k0, int lane) {
  int l = lane & 15, hi = lane >> 4;
  const _Float16* p = A + (size_t)(m0 + l) * lda + k0 + hi * 8;
  return make_frag(*(const h8*)p, *(const h8*)(p + 16));
}

// Same A tile, but source is fp32 with row index r = t*64+b decomposed as
// addr = (r&63)*cb + (r>>6)*ct + k  (converts to f16 in-register).
__device__ __forceinline__ v16h load_a_f32(const float* __restrict__ A, int cb, int ct,
                                           int m0, int k0, int lane) {
  int l = lane & 15, hi = lane >> 4;
  int r = m0 + l;
  const float* p = A + (size_t)(r & 63) * cb + (size_t)(r >> 6) * ct + k0 + hi * 8;
  f4 x0 = *(const f4*)p;        f4 x1 = *(const f4*)(p + 4);
  f4 x2 = *(const f4*)(p + 16); f4 x3 = *(const f4*)(p + 20);
  v16h a;
#pragma unroll
  for (int i = 0; i < 4; ++i) {
    a[i]      = (_Float16)x0[i];
    a[4 + i]  = (_Float16)x1[i];
    a[8 + i]  = (_Float16)x2[i];
    a[12 + i] = (_Float16)x3[i];
  }
  return a;
}

// B-matrix 32x16 f16 tile: B[k][n] = W[n][k], W row-major [*, ldw].
// lanes 0-15: col=lane, K chunks [k0+0..7] and [k0+16..23]; lanes16-31: col=lane-16, K chunks [k0+8..15] and [k0+24..31].
// Each lane reads two 8-half chunks of one weight row.
__device__ __forceinline__ v16h load_b(const _Float16* __restrict__ W, int ldw,
                                       int k0, int n0, int lane) {
  int l = lane & 15, hi = lane >> 4;
  const _Float16* p = W + (size_t)(n0 + l) * ldw + k0 + hi * 8;
  return make_frag(*(const h8*)p, *(const h8*)(p + 16));
}

__device__ __forceinline__ float sigmoid_fast(float x) {
  return 1.f / (1.f + __expf(-x));
}
__device__ __forceinline__ float tanh_fast(float x) {
  return 1.f - 2.f / (__expf(2.f * x) + 1.f);  // safe at +/-inf
}

// ---------------------------------------------------------------------------
// fp32 -> fp16 weight conversion
// ---------------------------------------------------------------------------
__global__ __launch_bounds__(256) void cvt_f32_f16(const float* __restrict__ src,
                                                   _Float16* __restrict__ dst, int n) {
  int i = (blockIdx.x * 256 + threadIdx.x) * 4;
  if (i < n) {
    f4 v = *(const f4*)(src + i);
    h4 o;
#pragma unroll
    for (int j = 0; j < 4; ++j) o[j] = (_Float16)v[j];
    *(h4*)(dst + i) = o;
  }
}

// ---------------------------------------------------------------------------
// Big parallel GEMM: out[r, g] = (f16) ( sum_k A[r,k] * W[g,k] + bias[g] )
// M = T*B = 32768, N = GG = 4096, K = 1024.
// One wave computes a 16x64 tile (A fragment reused across 4 N-subtiles).
// ---------------------------------------------------------------------------
template <bool AF16>
__global__ __launch_bounds__(256) void gemm_xproj(const void* __restrict__ Asrc,
                                                  int cb, int ct,
                                                  const _Float16* __restrict__ W,
                                                  const float* __restrict__ bias,
                                                  _Float16* __restrict__ out) {
  const int lane = threadIdx.x & 31;
  const int w    = blockIdx.x * 8 + (threadIdx.x >> 5);  // 0..131071
  const int m0   = (w >> 6) * 16;                         // 0..32752
  const int n0   = (w & 63) * 64;                         // 0..4032

  v8f acc0 = {}, acc1 = {}, acc2 = {}, acc3 = {};
  for (int k0 = 0; k0 < 1024; k0 += 32) {
    v16h a = AF16 ? load_a_f16((const _Float16*)Asrc, 1024, m0, k0, lane)
                  : load_a_f32((const float*)Asrc, cb, ct, m0, k0, lane);
    v16h b0 = load_b(W, 1024, k0, n0 +  0, lane);
    v16h b1 = load_b(W, 1024, k0, n0 + 16, lane);
    v16h b2 = load_b(W, 1024, k0, n0 + 32, lane);
    v16h b3 = load_b(W, 1024, k0, n0 + 48, lane);
    acc0 = __builtin_amdgcn_wmma_f32_16x16x32_f16(false, a, false, b0, (short)0, acc0, false, false);
    acc1 = __builtin_amdgcn_wmma_f32_16x16x32_f16(false, a, false, b1, (short)0, acc1, false, false);
    acc2 = __builtin_amdgcn_wmma_f32_16x16x32_f16(false, a, false, b2, (short)0, acc2, false, false);
    acc3 = __builtin_amdgcn_wmma_f32_16x16x32_f16(false, a, false, b3, (short)0, acc3, false, false);
  }

  const int l = lane & 15, hi = lane >> 4;
  v8f* accs[4] = {&acc0, &acc1, &acc2, &acc3};
#pragma unroll
  for (int s = 0; s < 4; ++s) {
    int col  = n0 + s * 16 + l;
    float bv = bias[col];
#pragma unroll
    for (int j = 0; j < 8; ++j) {
      int row = m0 + j + hi * 8;
      out[(size_t)row * GG + col] = (_Float16)((*accs[s])[j] + bv);
    }
  }
}

// ---------------------------------------------------------------------------
// Grid-wide barrier for the persistent scan kernel (16 workgroups).
// Start-value agnostic as long as counter starts at a multiple of nwg.
// ---------------------------------------------------------------------------
__device__ __forceinline__ void grid_barrier(unsigned* cnt, unsigned nwg) {
  __syncthreads();
  if (threadIdx.x == 0) {
    __threadfence();
    unsigned ticket = atomicAdd(cnt, 1u);
    unsigned target = (ticket / nwg + 1u) * nwg;
    while (*(volatile unsigned*)cnt < target) { __builtin_amdgcn_s_sleep(1); }
    __threadfence();
  }
  __syncthreads();
}

// ---------------------------------------------------------------------------
// Persistent LSTM scan: 16 WGs x 512 threads = 256 waves.
// Wave w owns rows m0..m0+15 (of B=64) x hidden cols h0..h0+15, across all 4
// gate blocks -> 4 WMMA accumulator chains + fused cell update in registers.
// Per step: z = xproj[t] + h_{t-1} @ Wh^T ; c,h update ; grid barrier.
// ---------------------------------------------------------------------------
__global__ __launch_bounds__(512) void lstm_scan(const _Float16* __restrict__ xproj, // [T*B, GG]
                                                 const _Float16* __restrict__ Wh,    // [GG, HH] f16
                                                 _Float16* __restrict__ hs,          // [T, B*HH] or null
                                                 _Float16* __restrict__ pp0,
                                                 _Float16* __restrict__ pp1,
                                                 float* __restrict__ out,            // [B*HH] or null
                                                 unsigned* __restrict__ barrier_cnt,
                                                 int use_hs) {
  const int lane = threadIdx.x & 31;
  const int w    = blockIdx.x * 16 + (threadIdx.x >> 5);  // 0..255
  const int m0   = (w >> 6) * 16;                          // row block (B)
  const int h0   = (w & 63) * 16;                          // hidden-unit block
  const int l    = lane & 15, hi = lane >> 4;
  const int col  = h0 + l;

  float c[8];
#pragma unroll
  for (int j = 0; j < 8; ++j) c[j] = 0.f;  // c_0 = 0, lives in VGPRs for all 512 steps

  for (int t = 0; t < TT; ++t) {
    v8f acc0 = {}, acc1 = {}, acc2 = {}, acc3 = {};
    if (t > 0) {
      const _Float16* hp = use_hs ? (hs + (size_t)(t - 1) * BB * HH)
                                  : (((t - 1) & 1) ? pp1 : pp0);
      for (int k0 = 0; k0 < HH; k0 += 32) {
        v16h a  = load_a_f16(hp, HH, m0, k0, lane);
        v16h b0 = load_b(Wh, HH, k0, 0 * HH + h0, lane);
        v16h b1 = load_b(Wh, HH, k0, 1 * HH + h0, lane);
        v16h b2 = load_b(Wh, HH, k0, 2 * HH + h0, lane);
        v16h b3 = load_b(Wh, HH, k0, 3 * HH + h0, lane);
        acc0 = __builtin_amdgcn_wmma_f32_16x16x32_f16(false, a, false, b0, (short)0, acc0, false, false);
        acc1 = __builtin_amdgcn_wmma_f32_16x16x32_f16(false, a, false, b1, (short)0, acc1, false, false);
        acc2 = __builtin_amdgcn_wmma_f32_16x16x32_f16(false, a, false, b2, (short)0, acc2, false, false);
        acc3 = __builtin_amdgcn_wmma_f32_16x16x32_f16(false, a, false, b3, (short)0, acc3, false, false);
      }
    }

    const _Float16* xp  = xproj + (size_t)t * BB * GG;
    _Float16* hout      = use_hs ? (hs + (size_t)t * BB * HH) : ((t & 1) ? pp1 : pp0);

#pragma unroll
    for (int j = 0; j < 8; ++j) {
      int row   = m0 + j + hi * 8;
      size_t zb = (size_t)row * GG + col;
      float zf = acc0[j] + (float)xp[zb + 0 * HH];  // forget
      float zw = acc1[j] + (float)xp[zb + 1 * HH];  // input gate
      float zi = acc2[j] + (float)xp[zb + 2 * HH];  // candidate
      float zm = acc3[j] + (float)xp[zb + 3 * HH];  // output gate
      c[j] = c[j] * sigmoid_fast(zf) + sigmoid_fast(zw) * tanh_fast(zi);
      float hval = tanh_fast(c[j]) * sigmoid_fast(zm);
      hout[(size_t)row * HH + col] = (_Float16)hval;
      if (out != nullptr && t == TT - 1) out[(size_t)row * HH + col] = hval;
    }
    grid_barrier(barrier_cnt, gridDim.x);  // publish h_t before anyone reads it at t+1
  }
}

// ---------------------------------------------------------------------------
// Host-side launch. Workspace layout (needs ~353 MB):
//   [0,       1MB)   : barrier counter (64B) + ping-pong h buffers (2 x 128KB)
//   [1MB,   33MB)    : fp16 weights Wx0f, Wh0f, Wx1f, Wh1f (8MB each)
//   [33MB, 289MB)    : xproj, f16 [T*B, 4096] (reused for layer 0 and layer 1)
//   [289MB,353MB)    : hs0,   f16 [T, B*H]
// ---------------------------------------------------------------------------
extern "C" void kernel_launch(void* const* d_in, const int* in_sizes, int n_in,
                              void* d_out, int out_size, void* d_ws, size_t ws_size,
                              hipStream_t stream) {
  const float* seq = (const float*)d_in[0];  // [B, T, I]
  const float* Wx0 = (const float*)d_in[1];  // [4H, I]
  const float* Wh0 = (const float*)d_in[2];  // [4H, H]
  const float* b0  = (const float*)d_in[3];  // [4H]
  const float* Wx1 = (const float*)d_in[4];  // [4H, H]
  const float* Wh1 = (const float*)d_in[5];  // [4H, H]
  const float* b1  = (const float*)d_in[6];  // [4H]
  float* out       = (float*)d_out;          // [B, H]

  uint8_t* ws = (uint8_t*)d_ws;
  unsigned*  cnt   = (unsigned*)ws;
  _Float16*  pp0   = (_Float16*)(ws + 1024);
  _Float16*  pp1   = pp0 + (size_t)BB * HH;
  _Float16*  wx0f  = (_Float16*)(ws + (1ull << 20));
  _Float16*  wh0f  = wx0f + (size_t)GG * II;
  _Float16*  wx1f  = wh0f + (size_t)GG * HH;
  _Float16*  wh1f  = wx1f + (size_t)GG * HH;
  _Float16*  xproj = (_Float16*)(ws + (1ull << 20) + 4ull * GG * 1024 * 2);
  _Float16*  hs0   = xproj + (size_t)TT * BB * GG;

  hipMemsetAsync(cnt, 0, 64, stream);  // graph-capture-safe barrier reset

  // fp32 -> fp16 weights (4M elements each)
  cvt_f32_f16<<<4096, 256, 0, stream>>>(Wx0, wx0f, GG * II);
  cvt_f32_f16<<<4096, 256, 0, stream>>>(Wh0, wh0f, GG * HH);
  cvt_f32_f16<<<4096, 256, 0, stream>>>(Wx1, wx1f, GG * HH);
  cvt_f32_f16<<<4096, 256, 0, stream>>>(Wh1, wh1f, GG * HH);

  // Layer 0 input projection: A = seq (fp32, addr = b*(T*I) + t*I + k)
  gemm_xproj<false><<<16384, 256, 0, stream>>>((const void*)seq, TT * II, II,
                                               wx0f, b0, xproj);
  // Layer 0 recurrent scan (persistent, writes hs0)
  lstm_scan<<<16, 512, 0, stream>>>(xproj, wh0f, hs0, nullptr, nullptr,
                                    nullptr, cnt, /*use_hs=*/1);
  // Layer 1 input projection: A = hs0 (fp16, addr = (t*64+b)*1024 + k)
  gemm_xproj<true><<<16384, 256, 0, stream>>>((const void*)hs0, HH, BB * HH,
                                              wx1f, b1, xproj);
  // Layer 1 recurrent scan (ping-pong h, emits final h to d_out at t=T-1)
  lstm_scan<<<16, 512, 0, stream>>>(xproj, wh1f, nullptr, pp0, pp1,
                                    out, cnt, /*use_hs=*/0);
}
